// Pnet2_model_6408091205909
// MI455X (gfx1250) — compile-verified, matrix operand order checked by reading, not run on hardware
//
#include <hip/hip_runtime.h>

// ---------------------------------------------------------------------------
// Types for CDNA5 WMMA (wave32, 16x16x32 bf16 -> f32 accum)
// ---------------------------------------------------------------------------
typedef __attribute__((ext_vector_type(16))) __bf16 v16bf;
typedef __attribute__((ext_vector_type(8)))  float  v8f;

__device__ __forceinline__ unsigned short f2bf(float f) {
    union { float f; unsigned int u; } v; v.f = f;
    unsigned int u = v.u;
    unsigned int r = (u + 0x7FFFu + ((u >> 16) & 1u)) >> 16;   // RNE
    return (unsigned short)r;
}
__device__ __forceinline__ float bf2f(unsigned short u) {
    union { unsigned int i; float f; } v; v.i = ((unsigned int)u) << 16;
    return v.f;
}

// ---------------------------------------------------------------------------
// Weight pre-pack: W f32 [N,K] row-major -> bf16 B-fragments laid out as
// [Ntiles][Kblocks][lane(32)][16 elems], zero padded in both N and K.
// Per ISA 7.12.2 (16-bit B 32x16): lane -> N=lane%16, K-half by lane>=16;
// element e -> K = kb*32 + (e/8)*16 + 8*(lane>=16) + e%8.
// ---------------------------------------------------------------------------
__global__ void pack_w(const float* __restrict__ W, unsigned short* __restrict__ out,
                       int N, int K, int Kb) {
    int t = blockIdx.x * blockDim.x + threadIdx.x;
    int total = ((N + 15) >> 4) * Kb * 32;
    if (t >= total) return;
    int lane = t & 31;
    int blk  = t >> 5;               // nt*Kb + kb
    int kb   = blk % Kb;
    int nt   = blk / Kb;
    int n    = nt * 16 + (lane & 15);
    int khalf = (lane >> 4) * 8;
    size_t ob = (size_t)t * 16;
#pragma unroll
    for (int e = 0; e < 16; ++e) {
        int k = kb * 32 + ((e >> 3) << 4) + khalf + (e & 7);
        out[ob + e] = (n < N && k < K) ? f2bf(W[(size_t)n * K + k]) : (unsigned short)0;
    }
}

// ---------------------------------------------------------------------------
// bf16 WMMA GEMM: out[M,N] = epilogue( A[M, 32*Kb](bf16, zero-padded) * B^T )
// A: row-major, lda multiple of 32 (16B-aligned fragments).
// Bp: pre-packed fragments (see pack_w).
// mode 0: f32 out, acc+bias | 1: bf16 out, relu(acc+bias)
// mode 2: bf16 out, relu((acc+bias)*scale+shift)
// One wave = one 16x16 tile. Hot loop: 4x b128 load + 1x wmma, no branches.
// ---------------------------------------------------------------------------
__global__ void gemm_wmma(const unsigned short* __restrict__ A, int lda,
                          const unsigned short* __restrict__ Bp,
                          const float* __restrict__ bias,
                          const float* __restrict__ scale,
                          const float* __restrict__ shift,
                          void* __restrict__ Out, int ldo,
                          int M, int Kb, int N, int mode) {
    const int Mtiles = (M + 15) >> 4;
    const int Ntiles = (N + 15) >> 4;
    const int wave   = (blockIdx.x * blockDim.x + threadIdx.x) >> 5;
    if (wave >= Mtiles * Ntiles) return;        // whole-wave uniform exit
    const int mt   = wave / Ntiles;
    const int nt   = wave % Ntiles;
    const int lane = threadIdx.x & 31;
    const int khalf = (lane >> 4) * 8;
    const unsigned short* Arow = A + (size_t)(mt * 16 + (lane & 15)) * lda + khalf;
    const unsigned short* Blane = Bp + ((size_t)nt * Kb * 32 + lane) * 16;

    v8f acc = {};
    for (int kb = 0; kb < Kb; ++kb) {
        union { v16bf v; uint4 q[2]; } a, b;
        a.q[0] = *(const uint4*)(Arow + kb * 32);
        a.q[1] = *(const uint4*)(Arow + kb * 32 + 16);
        b.q[0] = *(const uint4*)(Blane + (size_t)kb * 512);
        b.q[1] = *(const uint4*)(Blane + (size_t)kb * 512 + 8);
        acc = __builtin_amdgcn_wmma_f32_16x16x32_bf16(
            false, a.v, false, b.v, (short)0, acc, false, false);
    }

    const int col  = nt * 16 + (lane & 15);     // C/D: N = lane%16
    const float bc = (bias  && col < N) ? bias[col]  : 0.f;
    const float sc = (scale && col < N) ? scale[col] : 1.f;
    const float tc = (shift && col < N) ? shift[col] : 0.f;
#pragma unroll
    for (int r = 0; r < 8; ++r) {               // C/D: M = r + 8*(lane>=16)
        int row = mt * 16 + r + ((lane >= 16) ? 8 : 0);
        if (row >= M || col >= N) continue;
        float v = acc[r] + bc;
        if (mode == 2) v = v * sc + tc;
        if (mode >= 1) v = fmaxf(v, 0.f);
        if (mode == 0) ((float*)Out)[(size_t)row * ldo + col] = v;
        else ((unsigned short*)Out)[(size_t)row * ldo + col] = f2bf(v);
    }
}

// ---------------------------------------------------------------------------
// Farthest point sampling: one block per frame, iterative argmax in LDS.
// ---------------------------------------------------------------------------
__global__ void fps_kernel(const float* __restrict__ xyz, int* __restrict__ out,
                           int N, int S) {
    const int b   = blockIdx.x;
    const int tid = threadIdx.x;                 // 256 threads
    const float* p = xyz + (size_t)b * N * 3;
    __shared__ float dist[1024];
    __shared__ float rv[256];
    __shared__ int   ri[256];
    __shared__ int   s_far;
    for (int i = tid; i < N; i += 256) dist[i] = 1e10f;
    if (tid == 0) s_far = 0;
    __syncthreads();
    for (int it = 0; it < S; ++it) {
        int far = s_far;
        if (tid == 0) out[b * S + it] = far;
        float cx = p[far * 3 + 0], cy = p[far * 3 + 1], cz = p[far * 3 + 2];
        float bv = -1.f; int bi = 0;
        for (int i = tid; i < N; i += 256) {
            float dx = p[i * 3 + 0] - cx, dy = p[i * 3 + 1] - cy, dz = p[i * 3 + 2] - cz;
            float d  = fminf(dist[i], dx * dx + dy * dy + dz * dz);
            dist[i]  = d;
            if (d > bv) { bv = d; bi = i; }
        }
        rv[tid] = bv; ri[tid] = bi;
        __syncthreads();
        for (int off = 128; off > 0; off >>= 1) {
            if (tid < off) {
                if (rv[tid + off] > rv[tid] ||
                    (rv[tid + off] == rv[tid] && ri[tid + off] < ri[tid])) {
                    rv[tid] = rv[tid + off]; ri[tid] = ri[tid + off];
                }
            }
            __syncthreads();
        }
        if (tid == 0) s_far = ri[0];
        __syncthreads();
    }
}

__global__ void gather_xyz(const float* __restrict__ xyz, const int* __restrict__ idx,
                           float* __restrict__ out, int N, int S, int BT) {
    int g = blockIdx.x * blockDim.x + threadIdx.x;
    if (g >= BT * S) return;
    int bt = g / S;
    int i  = idx[g];
    out[(size_t)g * 3 + 0] = xyz[((size_t)bt * N + i) * 3 + 0];
    out[(size_t)g * 3 + 1] = xyz[((size_t)bt * N + i) * 3 + 1];
    out[(size_t)g * 3 + 2] = xyz[((size_t)bt * N + i) * 3 + 2];
}

// First nsample in-radius indices in index order; misses filled with first hit.
__global__ void ball_query_kernel(const float* __restrict__ xyz,
                                  const float* __restrict__ ctr,
                                  int* __restrict__ out,
                                  float r2, int N, int S, int K, int BT) {
    int g = blockIdx.x * blockDim.x + threadIdx.x;
    if (g >= BT * S) return;
    int bt = g / S;
    const float* p = xyz + (size_t)bt * N * 3;
    float cx = ctr[(size_t)g * 3 + 0], cy = ctr[(size_t)g * 3 + 1], cz = ctr[(size_t)g * 3 + 2];
    int cnt = 0, first = 0;
    for (int i = 0; i < N && cnt < K; ++i) {
        float dx = p[i * 3 + 0] - cx, dy = p[i * 3 + 1] - cy, dz = p[i * 3 + 2] - cz;
        if (dx * dx + dy * dy + dz * dz <= r2) {
            if (cnt == 0) first = i;
            out[(size_t)g * K + cnt] = i;
            ++cnt;
        }
    }
    for (; cnt < K; ++cnt) out[(size_t)g * K + cnt] = first;
}

// SA1 grouping: centered xyz -> bf16 rows [BT*S*K, 32] (K padded 3->32, zeros)
__global__ void group_sa1(const float* __restrict__ xyz, const float* __restrict__ nxyz,
                          const int* __restrict__ gidx, unsigned short* __restrict__ out,
                          int N, int S, int K, int BT) {
    int r = blockIdx.x * blockDim.x + threadIdx.x;
    if (r >= BT * S * K) return;
    int s  = (r / K) % S;
    int bt = r / (K * S);
    int i  = gidx[r];
    size_t ob = (size_t)r * 32;
#pragma unroll
    for (int c = 0; c < 3; ++c)
        out[ob + c] = f2bf(xyz[((size_t)bt * N + i) * 3 + c] -
                           nxyz[((size_t)bt * S + s) * 3 + c]);
    for (int c = 3; c < 32; ++c) out[ob + c] = 0;
}

// SA2 grouping: [centered xyz(3) | l1p feats(128) | pad(29)] -> bf16 rows of 160
__global__ void group_sa2(const float* __restrict__ l1x, const float* __restrict__ l2x,
                          const int* __restrict__ gidx, const unsigned short* __restrict__ l1p,
                          unsigned short* __restrict__ out,
                          int Nsrc, int S, int K, int C, int BT) {
    int r = blockIdx.x * blockDim.x + threadIdx.x;
    if (r >= BT * S * K) return;
    int s  = (r / K) % S;
    int bt = r / (K * S);
    int i  = gidx[r];
    size_t ob = (size_t)r * 160;
#pragma unroll
    for (int c = 0; c < 3; ++c)
        out[ob + c] = f2bf(l1x[((size_t)bt * Nsrc + i) * 3 + c] -
                           l2x[((size_t)bt * S + s) * 3 + c]);
    const unsigned short* src = l1p + ((size_t)bt * Nsrc + i) * C;
    for (int j = 0; j < C; ++j) out[ob + 3 + j] = src[j];
    for (int j = 3 + C; j < 160; ++j) out[ob + j] = 0;
}

// SA3 group_all concat: [l2x(3) | l2p(256) | pad(29)] -> bf16 rows of 288
__global__ void concat_sa3(const float* __restrict__ l2x, const unsigned short* __restrict__ l2p,
                           unsigned short* __restrict__ out, int rows) {
    int r = blockIdx.x * blockDim.x + threadIdx.x;
    if (r >= rows) return;
    size_t ob = (size_t)r * 288;
#pragma unroll
    for (int c = 0; c < 3; ++c) out[ob + c] = f2bf(l2x[(size_t)r * 3 + c]);
    const unsigned short* src = l2p + (size_t)r * 256;
    for (int j = 0; j < 256; ++j) out[ob + 3 + j] = src[j];
    for (int j = 259; j < 288; ++j) out[ob + j] = 0;
}

// max over K rows: in [G,K,C] -> out [G,C]   (bf16)
__global__ void maxk_kernel(const unsigned short* __restrict__ in,
                            unsigned short* __restrict__ out, int G, int K, int C) {
    int t = blockIdx.x * blockDim.x + threadIdx.x;
    if (t >= G * C) return;
    int g = t / C, c = t % C;
    float m = -3.0e38f;
    for (int k = 0; k < K; ++k)
        m = fmaxf(m, bf2f(in[((size_t)g * K + k) * C + c]));
    out[(size_t)t] = f2bf(m);
}

__global__ void zero_kernel(unsigned int* __restrict__ p, int nwords) {
    int t = blockIdx.x * blockDim.x + threadIdx.x;
    if (t < nwords) p[t] = 0u;
}

// GRU gate math + state update; writes hidden into y[b,t, dirOff + j] (bf16)
__global__ void gru_pointwise(const float* __restrict__ gi, const float* __restrict__ gh,
                              float* __restrict__ h, unsigned short* __restrict__ hbf,
                              unsigned short* __restrict__ y, int t, int dirOff) {
    int tid = blockIdx.x * blockDim.x + threadIdx.x;       // 8 * 512
    if (tid >= 8 * 512) return;
    int b = tid >> 9, j = tid & 511;
    float ir = gi[b * 1536 + j], iz = gi[b * 1536 + 512 + j], ig = gi[b * 1536 + 1024 + j];
    float hr = gh[b * 1536 + j], hz = gh[b * 1536 + 512 + j], hg = gh[b * 1536 + 1024 + j];
    float r = 1.f / (1.f + __expf(-(ir + hr)));
    float z = 1.f / (1.f + __expf(-(iz + hz)));
    float n = tanhf(ig + r * hg);
    float hv = (1.f - z) * n + z * h[tid];
    h[tid] = hv;
    hbf[tid] = f2bf(hv);
    y[((size_t)(b * 32 + t)) * 1024 + dirOff + j] = f2bf(hv);
}

// ---------------------------------------------------------------------------
// Host-side orchestration
// ---------------------------------------------------------------------------
extern "C" void kernel_launch(void* const* d_in, const int* in_sizes, int n_in,
                              void* d_out, int out_size, void* d_ws, size_t ws_size,
                              hipStream_t stream) {
    (void)in_sizes; (void)n_in; (void)out_size; (void)ws_size;
    const int BT = 256;                          // B*T frames
    auto F = [&](int i) { return (const float*)d_in[i]; };
    const float* xyz = F(0);                     // [8,32,1024,3]
    // params (insertion-order flattening): sa1@1, sa2@13, sa3@25, gru@37, fc@45
    const float *bih_f = F(39), *bhh_f = F(40), *bih_b = F(43), *bhh_b = F(44);
    const float *fc1b = F(46), *fc2b = F(48), *fc3b = F(50);

    // --- bump allocator over d_ws ---
    size_t off = 0;
    auto alloc = [&](size_t bytes) -> void* {
        void* p = (char*)d_ws + off;
        off += (bytes + 255) & ~(size_t)255;
        return p;
    };
    unsigned short* bufA = (unsigned short*)alloc((size_t)176 << 20);  // ping
    unsigned short* bufB = (unsigned short*)alloc((size_t)272 << 20);  // pong
    int*            fidx1 = (int*)alloc(32768u * 4);
    float*          l1x   = (float*)alloc(98304u * 4);
    int*            gidx1 = (int*)alloc(524288u * 4);
    unsigned short* l1p   = (unsigned short*)alloc(4194304u * 2);
    int*            fidx2 = (int*)alloc(16384u * 4);
    float*          l2x   = (float*)alloc(12288u * 4);
    int*            gidx2 = (int*)alloc(524288u * 4);
    unsigned short* l2p   = (unsigned short*)alloc(4194304u * 2);
    unsigned short* feats = (unsigned short*)alloc(524288u * 2);  // oversized: M=8 GEMM reads rows 8-15
    float*          gi    = (float*)alloc(8u * 1536 * 4);
    float*          gh    = (float*)alloc(8u * 1536 * 4);
    float*          hA    = (float*)alloc(4096u * 4);
    unsigned short* hbf   = (unsigned short*)alloc(8192u * 2);    // oversized for same reason
    unsigned short* ybuf  = (unsigned short*)alloc(262144u * 2);
    unsigned short* x1    = (unsigned short*)alloc(131072u * 2);
    unsigned short* x2    = (unsigned short*)alloc(65536u * 2);

    auto cdiv = [](int a, int b) { return (a + b - 1) / b; };

    // pre-pack all weights into WMMA B-fragment layout (bf16, zero padded)
    auto pack = [&](const float* W, int N, int K, int Kb) -> const unsigned short* {
        int Ntl = cdiv(N, 16);
        unsigned short* p = (unsigned short*)alloc((size_t)Ntl * Kb * 512 * 2);
        int total = Ntl * Kb * 32;
        pack_w<<<cdiv(total, 256), 256, 0, stream>>>(W, p, N, K, Kb);
        return p;
    };
    const unsigned short* Psa1_0 = pack(F(1),   64,    3,  1);
    const unsigned short* Psa1_1 = pack(F(5),   64,   64,  2);
    const unsigned short* Psa1_2 = pack(F(9),  128,   64,  2);
    const unsigned short* Psa2_0 = pack(F(13), 128,  131,  5);
    const unsigned short* Psa2_1 = pack(F(17), 128,  128,  4);
    const unsigned short* Psa2_2 = pack(F(21), 256,  128,  4);
    const unsigned short* Psa3_0 = pack(F(25), 256,  259,  9);
    const unsigned short* Psa3_1 = pack(F(29), 512,  256,  8);
    const unsigned short* Psa3_2 = pack(F(33), 1024, 512, 16);
    const unsigned short* PWih_f = pack(F(37), 1536, 1024, 32);
    const unsigned short* PWhh_f = pack(F(38), 1536,  512, 16);
    const unsigned short* PWih_b = pack(F(41), 1536, 1024, 32);
    const unsigned short* PWhh_b = pack(F(42), 1536,  512, 16);
    const unsigned short* PFc1   = pack(F(45),  512, 1024, 32);
    const unsigned short* PFc2   = pack(F(47),  256,  512, 16);
    const unsigned short* PFc3   = pack(F(49),   36,  256,  8);

    auto gemm = [&](const unsigned short* A, int lda, const unsigned short* Bp,
                    const float* b, const float* s, const float* t, void* out, int ldo,
                    int M, int Kb, int N, int mode) {
        int tiles  = cdiv(M, 16) * cdiv(N, 16);
        int blocks = cdiv(tiles * 32, 256);
        gemm_wmma<<<blocks, 256, 0, stream>>>(A, lda, Bp, b, s, t, out, ldo, M, Kb, N, mode);
    };

    // ===================== SA1 =====================
    fps_kernel<<<BT, 256, 0, stream>>>(xyz, fidx1, 1024, 128);
    gather_xyz<<<cdiv(BT * 128, 256), 256, 0, stream>>>(xyz, fidx1, l1x, 1024, 128, BT);
    ball_query_kernel<<<cdiv(BT * 128, 256), 256, 0, stream>>>(xyz, l1x, gidx1,
                                                               0.04f, 1024, 128, 16, BT);
    group_sa1<<<cdiv(524288, 256), 256, 0, stream>>>(xyz, l1x, gidx1, bufA, 1024, 128, 16, BT);
    gemm(bufA, 32, Psa1_0, F(2),  F(3),  F(4),  bufB,  64, 524288, 1,  64, 2);
    gemm(bufB, 64, Psa1_1, F(6),  F(7),  F(8),  bufA,  64, 524288, 2,  64, 2);
    gemm(bufA, 64, Psa1_2, F(10), F(11), F(12), bufB, 128, 524288, 2, 128, 2);
    maxk_kernel<<<cdiv(32768 * 128, 256), 256, 0, stream>>>(bufB, l1p, 32768, 16, 128);

    // ===================== SA2 =====================
    fps_kernel<<<BT, 256, 0, stream>>>(l1x, fidx2, 128, 64);
    gather_xyz<<<cdiv(BT * 64, 256), 256, 0, stream>>>(l1x, fidx2, l2x, 128, 64, BT);
    ball_query_kernel<<<cdiv(BT * 64, 256), 256, 0, stream>>>(l1x, l2x, gidx2,
                                                              0.16f, 128, 64, 32, BT);
    group_sa2<<<cdiv(524288, 256), 256, 0, stream>>>(l1x, l2x, gidx2, l1p, bufA,
                                                     128, 64, 32, 128, BT);
    gemm(bufA, 160, Psa2_0, F(14), F(15), F(16), bufB, 128, 524288, 5, 128, 2);
    gemm(bufB, 128, Psa2_1, F(18), F(19), F(20), bufA, 128, 524288, 4, 128, 2);
    gemm(bufA, 128, Psa2_2, F(22), F(23), F(24), bufB, 256, 524288, 4, 256, 2);
    maxk_kernel<<<cdiv(16384 * 256, 256), 256, 0, stream>>>(bufB, l2p, 16384, 32, 256);

    // ===================== SA3 (group_all) =====================
    concat_sa3<<<cdiv(16384, 256), 256, 0, stream>>>(l2x, l2p, bufA, 16384);
    gemm(bufA, 288, Psa3_0, F(26), F(27), F(28), bufB,  256, 16384,  9,  256, 2);
    gemm(bufB, 256, Psa3_1, F(30), F(31), F(32), bufA,  512, 16384,  8,  512, 2);
    gemm(bufA, 512, Psa3_2, F(34), F(35), F(36), bufB, 1024, 16384, 16, 1024, 2);
    maxk_kernel<<<cdiv(256 * 1024, 256), 256, 0, stream>>>(bufB, feats, 256, 64, 1024);

    // ===================== bidirectional GRU =====================
    zero_kernel<<<cdiv(4096, 256), 256, 0, stream>>>((unsigned int*)hA, 4096);
    zero_kernel<<<cdiv(4096, 256), 256, 0, stream>>>((unsigned int*)hbf, 4096);
    for (int t = 0; t < 32; ++t) {
        gemm(feats + (size_t)t * 1024, 32 * 1024, PWih_f, bih_f, nullptr, nullptr,
             gi, 1536, 8, 32, 1536, 0);
        gemm(hbf, 512, PWhh_f, bhh_f, nullptr, nullptr, gh, 1536, 8, 16, 1536, 0);
        gru_pointwise<<<16, 256, 0, stream>>>(gi, gh, hA, hbf, ybuf, t, 0);
    }
    zero_kernel<<<cdiv(4096, 256), 256, 0, stream>>>((unsigned int*)hA, 4096);
    zero_kernel<<<cdiv(4096, 256), 256, 0, stream>>>((unsigned int*)hbf, 4096);
    for (int t = 31; t >= 0; --t) {
        gemm(feats + (size_t)t * 1024, 32 * 1024, PWih_b, bih_b, nullptr, nullptr,
             gi, 1536, 8, 32, 1536, 0);
        gemm(hbf, 512, PWhh_b, bhh_b, nullptr, nullptr, gh, 1536, 8, 16, 1536, 0);
        gru_pointwise<<<16, 256, 0, stream>>>(gi, gh, hA, hbf, ybuf, t, 512);
    }

    // ===================== FC head =====================
    gemm(ybuf, 1024, PFc1, fc1b, nullptr, nullptr, x1, 512, 256, 32, 512, 1);
    gemm(x1,    512, PFc2, fc2b, nullptr, nullptr, x2, 256, 256, 16, 256, 1);
    gemm(x2,    256, PFc3, fc3b, nullptr, nullptr, d_out, 36, 256, 8, 36, 0);
}
